// ChatDecoder_54219667145328
// MI455X (gfx1250) — matrix operand order repaired and checked by reading, not compiled, not run on hardware
//
#include <hip/hip_runtime.h>
#include <hip/hip_bf16.h>

#define VOC    32000
#define EMBED  512
#define UNITS  1024
#define MAX_LEN 32
#define BATCH  64
#define INDICE_GO 1

typedef __attribute__((ext_vector_type(16))) __bf16 v16bf;
typedef __attribute__((ext_vector_type(8)))  float  v8f;
typedef __attribute__((ext_vector_type(4)))  unsigned int u32x4;

union FragU { u32x4 u4[2]; v16bf v; };

__device__ __forceinline__ unsigned short f2bf(float x) {
    unsigned int u = __float_as_uint(x);
    u += 0x7fffu + ((u >> 16) & 1u);          // round-to-nearest-even
    return (unsigned short)(u >> 16);
}

__device__ __forceinline__ float sigm(float x) {
    return 1.0f / (1.0f + __expf(-x));
}

// A fragment (16x32 bf16): base is lane-adjusted row pointer
//   base = A + row*ld + 16*(lane>>4);  chunk kc: 32 contiguous bytes at +kc
__device__ __forceinline__ v16bf load_a32(const unsigned short* base, int kc) {
    FragU f;
    const u32x4* q = reinterpret_cast<const u32x4*>(base + kc);
    f.u4[0] = q[0];
    f.u4[1] = q[1];
    return f.v;
}

// B fragment (32x16 bf16 from transposed Wt[N][K]): base is lane-adjusted
//   base = Wt + ncol*ld + 16*(lane>>4);  chunk kc: 32 contiguous bytes at +kc
__device__ __forceinline__ v16bf load_b32(const unsigned short* base, int kc) {
    FragU f;
    const u32x4* q = reinterpret_cast<const u32x4*>(base + kc);
    f.u4[0] = q[0];
    f.u4[1] = q[1];
    return f.v;
}

#define WMMA_BF16(A, B, C) \
    __builtin_amdgcn_wmma_f32_16x16x32_bf16(false, (A), false, (B), (short)0, (C), false, false)

// ---- prep: fp32 W[K][N] -> bf16 Wt[N][K] (one-time per call, L2-resident after)
__global__ __launch_bounds__(256) void transpose_to_bf16(
    const float* __restrict__ W, unsigned short* __restrict__ Wt, int K, int N) {
    __shared__ float tile[16][17];
    int n = blockIdx.x * 16 + threadIdx.x;
    int k = blockIdx.y * 16 + threadIdx.y;
    tile[threadIdx.y][threadIdx.x] = W[(size_t)k * N + n];
    __syncthreads();
    int nw = blockIdx.x * 16 + threadIdx.y;
    int kw = blockIdx.y * 16 + threadIdx.x;
    Wt[(size_t)nw * K + kw] = f2bf(tile[threadIdx.x][threadIdx.y]);
}

__global__ __launch_bounds__(256) void init_state(
    const float* __restrict__ h0, const float* __restrict__ c0,
    unsigned short* __restrict__ hb, float* __restrict__ c) {
    int i = blockIdx.x * blockDim.x + threadIdx.x;   // 64*1024 total
    hb[i] = f2bf(h0[i]);
    c[i]  = c0[i];
}

// ---- per step: argmax over prev logits (or GO at t=0) + bf16 embedding gather
__global__ __launch_bounds__(256) void argmax_embed(
    const float* __restrict__ out, const float* __restrict__ emb,
    unsigned short* __restrict__ xb, int t) {
    __shared__ float sv[256];
    __shared__ int   si[256];
    int b = blockIdx.x;
    int tid = threadIdx.x;
    int idx;
    if (t == 0) {
        idx = INDICE_GO;
    } else {
        const float* row = out + ((size_t)b * MAX_LEN + (t - 1)) * VOC;
        float best = -3.4e38f; int bi = 0;
        for (int j = tid; j < VOC; j += 256) {
            float v = row[j];
            if (v > best) { best = v; bi = j; }
        }
        sv[tid] = best; si[tid] = bi;
        __syncthreads();
        for (int s = 128; s > 0; s >>= 1) {
            if (tid < s) {
                if (sv[tid + s] > sv[tid] ||
                    (sv[tid + s] == sv[tid] && si[tid + s] < si[tid])) {
                    sv[tid] = sv[tid + s]; si[tid] = si[tid + s];
                }
            }
            __syncthreads();
        }
        idx = si[0];
    }
    const float* e = emb + (size_t)idx * EMBED;
    for (int j = tid; j < EMBED; j += 256)
        xb[b * EMBED + j] = f2bf(e[j]);
}

// ---- LSTM: each wave owns a 16x16 (m,n) h-tile and its 4 gate tiles.
// Two-stage pipelined K loop: fragments run one 32-chunk ahead of WMMAs.
__global__ __launch_bounds__(256) void lstm_step(
    const unsigned short* __restrict__ xb,     // [64][512]  bf16
    const unsigned short* __restrict__ hin,    // [64][1024] bf16
    const unsigned short* __restrict__ Wih_t,  // [4096][512]
    const unsigned short* __restrict__ Whh_t,  // [4096][1024]
    const float* __restrict__ bias,            // [4096]
    float* __restrict__ c,                     // [64][1024]
    unsigned short* __restrict__ hout) {       // [64][1024] bf16
    int lane = threadIdx.x & 31;
    int gw = blockIdx.x * (blockDim.x >> 5) + (threadIdx.x >> 5); // 0..255
    int m0 = (gw & 3) << 4;
    int n0 = (gw >> 2) << 4;
    int hi = lane >> 4;
    int lo = lane & 15;

    v8f acc[4];
#pragma unroll
    for (int g = 0; g < 4; ++g) {
        float bv = bias[(g << 10) + n0 + lo];
#pragma unroll
        for (int r = 0; r < 8; ++r) acc[g][r] = bv;
    }

    // lane-adjusted base pointers (hoisted; chunk offsets are immediates)
    const unsigned short* ax = xb  + (size_t)(m0 + lo) * EMBED + (hi << 4);
    const unsigned short* ah = hin + (size_t)(m0 + lo) * UNITS + (hi << 4);
    const unsigned short* bx[4];
    const unsigned short* bh[4];
#pragma unroll
    for (int g = 0; g < 4; ++g) {
        int ncol = (g << 10) + n0 + lo;
        bx[g] = Wih_t + (size_t)ncol * EMBED + (hi << 4);
        bh[g] = Whh_t + (size_t)ncol * UNITS + (hi << 4);
    }

    // ---- x @ W_ih (K = 512, 16 chunks, pipelined in pairs)
    {
        v16bf aP = load_a32(ax, 0);
        v16bf bP[4];
#pragma unroll
        for (int g = 0; g < 4; ++g) bP[g] = load_b32(bx[g], 0);
#pragma unroll
        for (int kc = 0; kc < EMBED; kc += 64) {
            v16bf aQ = load_a32(ax, kc + 32);
            v16bf bQ[4];
#pragma unroll
            for (int g = 0; g < 4; ++g) {
                bQ[g] = load_b32(bx[g], kc + 32);
                if (kc + 128 < EMBED) __builtin_prefetch(bx[g] + kc + 128, 0, 1);
            }
#pragma unroll
            for (int g = 0; g < 4; ++g) acc[g] = WMMA_BF16(aP, bP[g], acc[g]);
            if (kc + 64 < EMBED) {
                aP = load_a32(ax, kc + 64);
#pragma unroll
                for (int g = 0; g < 4; ++g) bP[g] = load_b32(bx[g], kc + 64);
            }
#pragma unroll
            for (int g = 0; g < 4; ++g) acc[g] = WMMA_BF16(aQ, bQ[g], acc[g]);
        }
    }

    // ---- h @ W_hh (K = 1024, 32 chunks, pipelined in pairs)
    {
        v16bf aP = load_a32(ah, 0);
        v16bf bP[4];
#pragma unroll
        for (int g = 0; g < 4; ++g) bP[g] = load_b32(bh[g], 0);
#pragma unroll
        for (int kc = 0; kc < UNITS; kc += 64) {
            v16bf aQ = load_a32(ah, kc + 32);
            v16bf bQ[4];
#pragma unroll
            for (int g = 0; g < 4; ++g) {
                bQ[g] = load_b32(bh[g], kc + 32);
                if (kc + 128 < UNITS) __builtin_prefetch(bh[g] + kc + 128, 0, 1);
            }
#pragma unroll
            for (int g = 0; g < 4; ++g) acc[g] = WMMA_BF16(aP, bP[g], acc[g]);
            if (kc + 64 < UNITS) {
                aP = load_a32(ah, kc + 64);
#pragma unroll
                for (int g = 0; g < 4; ++g) bP[g] = load_b32(bh[g], kc + 64);
            }
#pragma unroll
            for (int g = 0; g < 4; ++g) acc[g] = WMMA_BF16(aQ, bQ[g], acc[g]);
        }
    }

    // gates: element (r,lane) -> M = r + 8*(lane>>4), N = lane&15
#pragma unroll
    for (int r = 0; r < 8; ++r) {
        int M = r + (hi << 3);
        size_t off = (size_t)(m0 + M) * UNITS + n0 + lo;
        float iv = acc[0][r], fv = acc[1][r], gv = acc[2][r], ov = acc[3][r];
        float cp = c[off];
        float cn = sigm(fv) * cp + sigm(iv) * tanhf(gv);
        float hn = sigm(ov) * tanhf(cn);
        c[off] = cn;
        hout[off] = f2bf(hn);
    }
}

// ---- logits: each wave owns 4 N-tiles (64 cols) at one m-tile; A frag shared.
__global__ __launch_bounds__(256) void logits_step(
    const unsigned short* __restrict__ h,      // [64][1024] bf16 (new h)
    const unsigned short* __restrict__ Wd_t,   // [32000][1024]
    const float* __restrict__ bd,              // [32000]
    float* __restrict__ out, int t) {
    int lane = threadIdx.x & 31;
    int gw = blockIdx.x * (blockDim.x >> 5) + (threadIdx.x >> 5); // 0..1999
    int m0 = (gw & 3) << 4;
    int n0 = (gw >> 2) << 6;                   // 500 groups of 64 cols
    int hi = lane >> 4;
    int lo = lane & 15;

    v8f acc[4];
#pragma unroll
    for (int j = 0; j < 4; ++j) {
        float bv = bd[n0 + (j << 4) + lo];
#pragma unroll
        for (int r = 0; r < 8; ++r) acc[j][r] = bv;
    }

    const unsigned short* ab = h + (size_t)(m0 + lo) * UNITS + (hi << 4);
    const unsigned short* bb[4];
#pragma unroll
    for (int j = 0; j < 4; ++j)
        bb[j] = Wd_t + (size_t)(n0 + (j << 4) + lo) * UNITS + (hi << 4);

    v16bf aP = load_a32(ab, 0);
    v16bf bP[4];
#pragma unroll
    for (int j = 0; j < 4; ++j) bP[j] = load_b32(bb[j], 0);

#pragma unroll
    for (int kc = 0; kc < UNITS; kc += 64) {
        v16bf aQ = load_a32(ab, kc + 32);
        v16bf bQ[4];
#pragma unroll
        for (int j = 0; j < 4; ++j) {
            bQ[j] = load_b32(bb[j], kc + 32);
            if (kc + 128 < UNITS) __builtin_prefetch(bb[j] + kc + 128, 0, 1);
        }
#pragma unroll
        for (int j = 0; j < 4; ++j) acc[j] = WMMA_BF16(aP, bP[j], acc[j]);
        if (kc + 64 < UNITS) {
            aP = load_a32(ab, kc + 64);
#pragma unroll
            for (int j = 0; j < 4; ++j) bP[j] = load_b32(bb[j], kc + 64);
        }
#pragma unroll
        for (int j = 0; j < 4; ++j) acc[j] = WMMA_BF16(aQ, bQ[j], acc[j]);
    }

#pragma unroll
    for (int j = 0; j < 4; ++j) {
#pragma unroll
        for (int r = 0; r < 8; ++r) {
            int M = r + (hi << 3);
            size_t o = ((size_t)(m0 + M) * MAX_LEN + t) * VOC + n0 + (j << 4) + lo;
            out[o] = acc[j][r];
        }
    }
}

extern "C" void kernel_launch(void* const* d_in, const int* in_sizes, int n_in,
                              void* d_out, int out_size, void* d_ws, size_t ws_size,
                              hipStream_t stream) {
    (void)in_sizes; (void)n_in; (void)out_size; (void)ws_size;
    const float* h0      = (const float*)d_in[0];
    const float* c0      = (const float*)d_in[1];
    const float* emb     = (const float*)d_in[2];
    const float* W_ih    = (const float*)d_in[3];
    const float* W_hh    = (const float*)d_in[4];
    const float* b       = (const float*)d_in[5];
    const float* W_dense = (const float*)d_in[6];
    const float* b_dense = (const float*)d_in[7];
    float* out = (float*)d_out;

    // workspace layout (bf16 stored as ushort); total ~78.7 MB
    unsigned short* Wih_t = (unsigned short*)d_ws;                  // 4096*512
    unsigned short* Whh_t = Wih_t + (size_t)4096 * 512;             // 4096*1024
    unsigned short* Wd_t  = Whh_t + (size_t)4096 * 1024;            // 32000*1024
    unsigned short* hb0   = Wd_t  + (size_t)VOC * UNITS;            // 64*1024
    unsigned short* hb1   = hb0   + (size_t)BATCH * UNITS;
    unsigned short* xb    = hb1   + (size_t)BATCH * UNITS;          // 64*512
    float*          c     = (float*)(xb + (size_t)BATCH * EMBED);   // 64*1024 f32

    // one-time prep (re-done each call for determinism)
    transpose_to_bf16<<<dim3(4096 / 16, EMBED / 16), dim3(16, 16), 0, stream>>>(
        W_ih, Wih_t, EMBED, 4096);
    transpose_to_bf16<<<dim3(4096 / 16, UNITS / 16), dim3(16, 16), 0, stream>>>(
        W_hh, Whh_t, UNITS, 4096);
    transpose_to_bf16<<<dim3(VOC / 16, UNITS / 16), dim3(16, 16), 0, stream>>>(
        W_dense, Wd_t, UNITS, VOC);
    init_state<<<(BATCH * UNITS) / 256, 256, 0, stream>>>(h0, c0, hb0, c);

    for (int t = 0; t < MAX_LEN; ++t) {
        const unsigned short* hprev = (t & 1) ? hb1 : hb0;
        unsigned short*       hnew  = (t & 1) ? hb0 : hb1;
        argmax_embed<<<BATCH, 256, 0, stream>>>(out, emb, xb, t);
        lstm_step<<<32, 256, 0, stream>>>(xb, hprev, Wih_t, Whh_t, b, c, hnew);
        logits_step<<<250, 256, 0, stream>>>(hnew, Wd_t, b_dense, out, t);
    }
}
